// SpeNet_76398878261345
// MI455X (gfx1250) — matrix operand order repaired and checked by reading, not compiled, not run on hardware
//
#include <hip/hip_runtime.h>
#include <hip/hip_bf16.h>
#include <cstdint>

typedef __attribute__((ext_vector_type(16))) _Float16 v16h;
typedef __attribute__((ext_vector_type(8)))  _Float16 v8h;
typedef __attribute__((ext_vector_type(8)))  float    v8f;
typedef int v4i __attribute__((vector_size(16)));

#define BSZ 2048
#define CSTR 96     // channel slots per batch in X / T1
#define LSTR 512    // L stride (max L)
#define WBLOB 9216  // halves per packed weight blob (96 rows x 96 cols f16)

__device__ __forceinline__ v8f wmma16x16x32(v16h a, v16h b, v8f c) {
  return __builtin_amdgcn_wmma_f32_16x16x32_f16(false, a, false, b, (short)0, c, false, false);
}

// A fragment: lane (grp = lane>>4) reads two contiguous 8-half runs (ISA 7.12.2)
__device__ __forceinline__ v16h ld_frag_a(const _Float16* row, int kbase, int grp) {
  v8h a0 = *(const v8h*)(row + kbase + 8 * grp);
  v8h a1 = *(const v8h*)(row + kbase + 16 + 8 * grp);
  v16h r;
#pragma unroll
  for (int j = 0; j < 8; ++j) { r[j] = a0[j]; r[j + 8] = a1[j]; }
  return r;
}

// B fragment from TRANSPOSED storage Bt[n][K]: two contiguous 8-half runs (same k order as A)
__device__ __forceinline__ v16h ld_frag_b(const _Float16* col, int kbase, int grp) {
  v8h b0 = *(const v8h*)(col + kbase + 8 * grp);
  v8h b1 = *(const v8h*)(col + kbase + 16 + 8 * grp);
  v16h r;
#pragma unroll
  for (int j = 0; j < 8; ++j) { r[j] = b0[j]; r[j + 8] = b1[j]; }
  return r;
}

// ---- async global->LDS copy (gfx1250 GLOBAL_LOAD_ASYNC_TO_LDS_B128), guarded
#if __has_builtin(__builtin_amdgcn_global_load_async_to_lds_b128) && \
    __has_builtin(__builtin_amdgcn_s_wait_asynccnt)
#define HAVE_ASYNC_LDS 1
#else
#define HAVE_ASYNC_LDS 0
#endif

__device__ __forceinline__ void copy_b128_to_lds(const _Float16* g, _Float16* l) {
#if HAVE_ASYNC_LDS
  __builtin_amdgcn_global_load_async_to_lds_b128(
      (__attribute__((address_space(1))) v4i*)(uintptr_t)g,
      (__attribute__((address_space(3))) v4i*)(uint32_t)(uintptr_t)l,
      0, 0);
#else
  *(v8h*)l = *(const v8h*)g;
#endif
}
__device__ __forceinline__ void async_copy_wait() {
#if HAVE_ASYNC_LDS
  __builtin_amdgcn_s_wait_asynccnt(0);
#endif
}

// --------------------------------------------- pack fp32 weights -> f16 blob
// taps==1: dst[o*96+c] = W[o*Cin+c]; taps==3: dst[(tap*32+o)*96+c] = W[(o*Cin+c)*3+tap]
__global__ __launch_bounds__(256)
void pack_w_kernel(const float* __restrict__ W, int Ochs, int Cin, int taps,
                   _Float16* __restrict__ dst) {
  for (int idx = threadIdx.x + blockIdx.x * 256; idx < WBLOB; idx += gridDim.x * 256) {
    float val = 0.f;
    if (taps == 1) {
      int o = idx / 96, c = idx % 96;
      if (o < Ochs && c < Cin) val = W[o * Cin + c];
    } else {
      int tap = idx / (32 * 96), r = idx % (32 * 96);
      int o = r / 96, c = r % 96;
      if (o < Ochs && c < Cin) val = W[(o * Cin + c) * 3 + tap];
    }
    dst[idx] = (_Float16)val;
  }
}

// ---------------------------------------------------------------- stem conv
__global__ __launch_bounds__(256)
void stem_kernel(const float* __restrict__ xspe, const float* __restrict__ w,
                 _Float16* __restrict__ X) {
  int b = blockIdx.x;
  const float* xr = xspe + (size_t)b * 512;
  for (int idx = threadIdx.x; idx < 44 * 512; idx += 256) {
    int o = idx >> 9, l = idx & 511;
    float xm1 = (l > 0)   ? xr[l - 1] : 0.f;
    float x0  = xr[l];
    float xp1 = (l < 511) ? xr[l + 1] : 0.f;
    float y = w[o * 3 + 0] * xm1 + w[o * 3 + 1] * x0 + w[o * 3 + 2] * xp1;
    X[((size_t)b * CSTR + o) * LSTR + l] = (_Float16)y;
  }
}

// --------------------------------------------- BN+ReLU then 1x1 conv (GEMM)
__global__ __launch_bounds__(256)
void bnconv1x1_kernel(const _Float16* __restrict__ Xin, int Cin, int Llen,
                      const float* __restrict__ g, const float* __restrict__ bb,
                      const float* __restrict__ m, const float* __restrict__ v,
                      const _Float16* __restrict__ Wp, int Ochs,
                      _Float16* __restrict__ Out, int Ooff) {
  int ntl = Llen >> 6;
  int b = blockIdx.x / ntl, lt = blockIdx.x % ntl;
  int l0 = lt * 64;
  int Cp = (Cin + 31) & ~31;
  int Op = (Ochs + 15) & ~15;
  __shared__ __align__(16) _Float16 Asm[96 * 96];  // [o][c], stride 96
  __shared__ __align__(16) _Float16 Bt[64 * 96];   // [l][c], stride 96
  __shared__ float af[96], df[96];
  // A: async DMA of the pre-packed f16 blob
  for (int idx = threadIdx.x; idx < Op * 12; idx += 256)
    copy_b128_to_lds(Wp + idx * 8, Asm + idx * 8);
  if (threadIdx.x < (unsigned)Cp) {
    int c = threadIdx.x;
    float a = 0.f, d = 0.f;
    if (c < Cin) { a = g[c] * rsqrtf(v[c] + 1e-5f); d = bb[c] - m[c] * a; }
    af[c] = a; df[c] = d;
  }
  __syncthreads();
  // B: vectorized global read, BN+ReLU, transposed LDS write
  for (int idx = threadIdx.x; idx < Cp * 8; idx += 256) {
    int c = idx >> 3, seg = idx & 7;
    _Float16 o8[8];
    if (c < Cin) {
      v8h xx = *(const v8h*)(Xin + ((size_t)b * CSTR + c) * LSTR + l0 + seg * 8);
#pragma unroll
      for (int j = 0; j < 8; ++j)
        o8[j] = (_Float16)fmaxf(af[c] * (float)xx[j] + df[c], 0.f);
    } else {
#pragma unroll
      for (int j = 0; j < 8; ++j) o8[j] = (_Float16)0.f;
    }
#pragma unroll
    for (int j = 0; j < 8; ++j) Bt[(seg * 8 + j) * 96 + c] = o8[j];
  }
  async_copy_wait();
  __syncthreads();
  int wave = threadIdx.x >> 5, lane = threadIdx.x & 31;
  int grp = lane >> 4, lm = lane & 15;
  int tiles = (Op >> 4) * 4;
  for (int t = wave; t < tiles; t += 8) {
    int mt = t >> 2, nt = t & 3;
    const _Float16* arow = Asm + (mt * 16 + lm) * 96;
    const _Float16* bcol = Bt + (nt * 16 + lm) * 96;
    v8f acc = {};
    for (int ks = 0; ks < (Cp >> 5); ++ks) {
      v16h av = ld_frag_a(arow, ks * 32, grp);
      v16h bv = ld_frag_b(bcol, ks * 32, grp);
      acc = wmma16x16x32(av, bv, acc);
    }
#pragma unroll
    for (int r = 0; r < 8; ++r) {
      int mm = mt * 16 + r + 8 * grp, nn = nt * 16 + lm;
      if (mm < Ochs)
        Out[((size_t)b * CSTR + Ooff + mm) * LSTR + l0 + nn] = (_Float16)acc[r];
    }
  }
}

// --------------------------------- BN+ReLU then k=3 conv (3 shifted GEMMs)
__global__ __launch_bounds__(256)
void bnconv3_kernel(const _Float16* __restrict__ Xin, int Llen,
                    const float* __restrict__ g, const float* __restrict__ bb,
                    const float* __restrict__ m, const float* __restrict__ v,
                    const _Float16* __restrict__ Wp,  // packed [tap*32+o][96]
                    _Float16* __restrict__ Out) {     // [b][32][512]
  const int Cin = 88, Cp = 96, Ochs = 22;
  int ntl = Llen >> 6;
  int b = blockIdx.x / ntl, lt = blockIdx.x % ntl;
  int l0 = lt * 64;
  __shared__ __align__(16) _Float16 Asm[3 * 32 * 96];  // [tap][o][c]
  __shared__ __align__(16) _Float16 Bt[68 * 96];       // [haloCol][c]
  __shared__ float af[96], df[96];
  for (int idx = threadIdx.x; idx < WBLOB / 8; idx += 256)
    copy_b128_to_lds(Wp + idx * 8, Asm + idx * 8);
  if (threadIdx.x < (unsigned)Cp) {
    int c = threadIdx.x;
    float a = 0.f, d = 0.f;
    if (c < Cin) { a = g[c] * rsqrtf(v[c] + 1e-5f); d = bb[c] - m[c] * a; }
    af[c] = a; df[c] = d;
  }
  __syncthreads();
  for (int idx = threadIdx.x; idx < Cp * 68; idx += 256) {
    int c = idx / 68, col = idx % 68;
    int gl = l0 + col - 1;
    float val = 0.f;
    if (c < Cin && col < 66 && gl >= 0 && gl < Llen) {
      float xx = (float)Xin[((size_t)b * CSTR + c) * LSTR + gl];
      val = fmaxf(af[c] * xx + df[c], 0.f);
    }
    Bt[col * 96 + c] = (_Float16)val;
  }
  async_copy_wait();
  __syncthreads();
  int wave = threadIdx.x >> 5, lane = threadIdx.x & 31;
  int grp = lane >> 4, lm = lane & 15;
  int mt = wave >> 2, nt = wave & 3;  // 8 tiles, one per wave
  const _Float16* arowBase = Asm + (mt * 16 + lm) * 96;
  v8f acc = {};
  for (int tap = 0; tap < 3; ++tap) {
    const _Float16* arow = arowBase + tap * 32 * 96;
    const _Float16* bcol = Bt + (nt * 16 + lm + tap) * 96;
#pragma unroll
    for (int ks = 0; ks < 3; ++ks) {
      v16h av = ld_frag_a(arow, ks * 32, grp);
      v16h bv = ld_frag_b(bcol, ks * 32, grp);
      acc = wmma16x16x32(av, bv, acc);
    }
  }
#pragma unroll
  for (int r = 0; r < 8; ++r) {
    int mm = mt * 16 + r + 8 * grp, nn = nt * 16 + lm;
    if (mm < Ochs)
      Out[((size_t)b * 32 + mm) * LSTR + l0 + nn] = (_Float16)acc[r];
  }
}

// ------------------------------------- channel self-attention (per batch b)
// Dynamic LDS: xh[32*512] f16 | xt[512*32] f16 | simf[32*32] f32 | ath[32*32] f16 | invn[32]
#define ATT_SMEM (32 * 512 * 2 + 512 * 32 * 2 + 32 * 32 * 4 + 32 * 32 * 2 + 32 * 4)
__global__ __launch_bounds__(256)
void atten_kernel(const _Float16* __restrict__ T2, int Llen,
                  _Float16* __restrict__ X, int coff) {
  extern __shared__ __align__(16) char smem[];
  _Float16* xh   = (_Float16*)smem;             // [c][l]
  _Float16* xt   = xh + 32 * 512;               // [l][c]
  float*    simf = (float*)(xt + 512 * 32);     // raw gram
  _Float16* ath  = (_Float16*)(simf + 32 * 32); // softmaxed attn
  float*    invn = (float*)(ath + 32 * 32);
  int b = blockIdx.x;
  for (int idx = threadIdx.x; idx < 32 * 64; idx += 256) {
    int c = idx >> 6, ls = idx & 63;
    v8h val;
    if (c < 22 && ls * 8 < Llen) {
      val = *(const v8h*)(T2 + ((size_t)b * 32 + c) * LSTR + ls * 8);
    } else {
#pragma unroll
      for (int j = 0; j < 8; ++j) val[j] = (_Float16)0.f;
    }
    *(v8h*)(xh + (c << 9) + ls * 8) = val;
  }
  __syncthreads();
  // transpose copy (vectorized reads) + row norms
  for (int idx = threadIdx.x; idx < 32 * 64; idx += 256) {
    int c = idx & 31, ls = idx >> 5;
    v8h xx = *(const v8h*)(xh + (c << 9) + ls * 8);
#pragma unroll
    for (int j = 0; j < 8; ++j) xt[(ls * 8 + j) * 32 + c] = xx[j];
  }
  if (threadIdx.x < 32) {
    int r = threadIdx.x;
    float s = 0.f;
    for (int l = 0; l < Llen; l += 8) {
      v8h xv = *(const v8h*)(xh + (r << 9) + l);
#pragma unroll
      for (int j = 0; j < 8; ++j) { float f = (float)xv[j]; s += f * f; }
    }
    invn[r] = 1.f / fmaxf(sqrtf(s), 1e-8f);
  }
  __syncthreads();
  int wave = threadIdx.x >> 5, lane = threadIdx.x & 31;
  int grp = lane >> 4, lm = lane & 15;
  // raw gram G = x @ x^T (normalization folded into softmax pass)
  if (wave < 4) {
    int mt = wave >> 1, nt = wave & 1;
    const _Float16* arow = xh + (mt * 16 + lm) * 512;
    const _Float16* bcol = xh + (nt * 16 + lm) * 512;
    v8f acc = {};
    for (int ks = 0; ks < (Llen >> 5); ++ks) {
      v16h av = ld_frag_a(arow, ks * 32, grp);
      v16h bv = ld_frag_b(bcol, ks * 32, grp);
      acc = wmma16x16x32(av, bv, acc);
    }
#pragma unroll
    for (int r = 0; r < 8; ++r)
      simf[(mt * 16 + r + 8 * grp) * 32 + nt * 16 + lm] = acc[r];
  }
  __syncthreads();
  if (threadIdx.x < 32) {
    int r = threadIdx.x;
    if (r < 22) {
      float ir = invn[r];
      float mx = -1e30f;
      for (int c = 0; c < 22; ++c) mx = fmaxf(mx, simf[r * 32 + c] * ir * invn[c]);
      float s = 0.f;
      for (int c = 0; c < 22; ++c) s += __expf(simf[r * 32 + c] * ir * invn[c] - mx);
      float is = 1.f / s;
      for (int c = 0; c < 32; ++c)
        ath[r * 32 + c] =
            (_Float16)((c < 22) ? __expf(simf[r * 32 + c] * ir * invn[c] - mx) * is : 0.f);
    } else {
      for (int c = 0; c < 32; ++c) ath[r * 32 + c] = (_Float16)0.f;
    }
  }
  __syncthreads();
  // out = attn @ x + x  (K=32, one WMMA per tile; B from transposed xt)
  int ntl = Llen >> 4;
  for (int t = wave; t < 2 * ntl; t += 8) {
    int mt = t / ntl, nt = t % ntl;
    v16h av = ld_frag_a(ath + (mt * 16 + lm) * 32, 0, grp);
    v16h bv = ld_frag_b(xt + (nt * 16 + lm) * 32, 0, grp);
    v8f acc = {};
    acc = wmma16x16x32(av, bv, acc);
#pragma unroll
    for (int r = 0; r < 8; ++r) {
      int mm = mt * 16 + r + 8 * grp, nn = nt * 16 + lm;
      if (mm < 22) {
        float o = acc[r] + (float)xh[mm * 512 + nn];
        X[((size_t)b * CSTR + coff + mm) * LSTR + nn] = (_Float16)o;
      }
    }
  }
}

// --------------------------------------------------------- avg-pool by 2
__global__ __launch_bounds__(256)
void pool2_kernel(const _Float16* __restrict__ T1, int Lout,
                  _Float16* __restrict__ X) {
  int b = blockIdx.x;
  int nseg = Lout >> 3;
  for (int idx = threadIdx.x; idx < 44 * nseg; idx += 256) {
    int c = idx / nseg, sg = idx % nseg;
    const _Float16* src = T1 + ((size_t)b * CSTR + c) * LSTR + sg * 16;
    v8h a0 = *(const v8h*)(src);
    v8h a1 = *(const v8h*)(src + 8);
    v8h o;
#pragma unroll
    for (int j = 0; j < 4; ++j) {
      o[j]     = (_Float16)(0.5f * ((float)a0[2 * j] + (float)a0[2 * j + 1]));
      o[4 + j] = (_Float16)(0.5f * ((float)a1[2 * j] + (float)a1[2 * j + 1]));
    }
    *(v8h*)(X + ((size_t)b * CSTR + c) * LSTR + sg * 8) = o;
  }
}

// --------------------------------------------- head: BN+ReLU, mean, FC
__global__ __launch_bounds__(64)
void head_kernel(const _Float16* __restrict__ X,
                 const float* __restrict__ g, const float* __restrict__ bb,
                 const float* __restrict__ m, const float* __restrict__ v,
                 const float* __restrict__ fcw, const float* __restrict__ fcb,
                 float* __restrict__ out) {
  int b = blockIdx.x;
  __shared__ float ms[44];
  int t = threadIdx.x;
  if (t < 44) {
    float a = g[t] * rsqrtf(v[t] + 1e-5f);
    float d = bb[t] - m[t] * a;
    float s = 0.f;
    for (int l = 0; l < 64; l += 8) {
      v8h xx = *(const v8h*)(X + ((size_t)b * CSTR + t) * LSTR + l);
#pragma unroll
      for (int j = 0; j < 8; ++j) s += fmaxf(a * (float)xx[j] + d, 0.f);
    }
    ms[t] = s * (1.f / 64.f);
  }
  __syncthreads();
  if (t < 32) {
    float s = fcb[t];
    for (int c = 0; c < 44; ++c) s += ms[c] * fcw[c * 32 + t];
    out[(size_t)b * 32 + t] = s;
  }
}

// ------------------------------------------------------------------- host
struct BNp { const float *g, *b, *m, *v; };

extern "C" void kernel_launch(void* const* d_in, const int* in_sizes, int n_in,
                              void* d_out, int out_size, void* d_ws, size_t ws_size,
                              hipStream_t stream) {
  auto F = [&](int i) { return (const float*)d_in[i]; };
  const float *xspe, *conv1, *fcw, *fcb;
  BNp bnf, bn1[3][2], bn2[3][2], bnt[3];
  const float *w1[3][2], *w2[3][2], *wt[3];

  if (n_in > 0 && in_sizes[0] == 2048 * 512) {
    // dict insertion order flattening: x_spe first, bn dicts as (g,b,m,v)
    xspe = F(0); conv1 = F(1);
    for (int s = 0; s < 3; ++s) {
      int base = 2 + 25 * s;
      for (int bl = 0; bl < 2; ++bl) {
        int k = base + 10 * bl;
        bn1[s][bl] = { F(k + 0), F(k + 1), F(k + 2), F(k + 3) };
        w1[s][bl]  = F(k + 4);
        bn2[s][bl] = { F(k + 5), F(k + 6), F(k + 7), F(k + 8) };
        w2[s][bl]  = F(k + 9);
      }
      bnt[s] = { F(base + 20), F(base + 21), F(base + 22), F(base + 23) };
      wt[s]  = F(base + 24);
    }
    bnf = { F(77), F(78), F(79), F(80) };
    fcw = F(81); fcb = F(82);
  } else {
    // JAX pytree (sorted dict keys): params first, bn dicts as (b,g,m,v)
    bnf = { F(1), F(0), F(2), F(3) };
    conv1 = F(4); fcb = F(5); fcw = F(6);
    for (int s = 0; s < 3; ++s) {
      int base = 7 + 25 * s;
      for (int bl = 0; bl < 2; ++bl) {
        int k = base + 10 * bl;
        bn1[s][bl] = { F(k + 1), F(k + 0), F(k + 2), F(k + 3) };
        bn2[s][bl] = { F(k + 5), F(k + 4), F(k + 6), F(k + 7) };
        w1[s][bl]  = F(k + 8);
        w2[s][bl]  = F(k + 9);
      }
      bnt[s] = { F(base + 21), F(base + 20), F(base + 22), F(base + 23) };
      wt[s]  = F(base + 24);
    }
    xspe = F(82);
  }

  char* ws = (char*)d_ws;
  const size_t actBytes = (size_t)BSZ * CSTR * LSTR * sizeof(_Float16);
  const size_t t2Bytes  = (size_t)BSZ * 32 * LSTR * sizeof(_Float16);
  _Float16* X  = (_Float16*)ws;
  _Float16* T1 = (_Float16*)(ws + actBytes);
  _Float16* T2 = (_Float16*)(ws + 2 * actBytes);
  _Float16* WP = (_Float16*)(ws + 2 * actBytes + t2Bytes);

  // pack all conv weights to padded f16 blobs (blob i at WP + i*WBLOB)
  for (int s = 0; s < 3; ++s) {
    for (int bl = 0; bl < 2; ++bl) {
      pack_w_kernel<<<4, 256, 0, stream>>>(w1[s][bl], 88, bl ? 66 : 44, 1,
                                           WP + (size_t)(s * 2 + bl) * WBLOB);
      pack_w_kernel<<<4, 256, 0, stream>>>(w2[s][bl], 22, 88, 3,
                                           WP + (size_t)(6 + s * 2 + bl) * WBLOB);
    }
    pack_w_kernel<<<4, 256, 0, stream>>>(wt[s], 44, 88, 1,
                                         WP + (size_t)(12 + s) * WBLOB);
  }

  stem_kernel<<<BSZ, 256, 0, stream>>>(xspe, conv1, X);

  int L = 512;
  for (int s = 0; s < 3; ++s) {
    int inc = 44;
    for (int bl = 0; bl < 2; ++bl) {
      int ntl = L / 64;
      bnconv1x1_kernel<<<BSZ * ntl, 256, 0, stream>>>(
          X, inc, L, bn1[s][bl].g, bn1[s][bl].b, bn1[s][bl].m, bn1[s][bl].v,
          WP + (size_t)(s * 2 + bl) * WBLOB, 88, T1, 0);
      bnconv3_kernel<<<BSZ * ntl, 256, 0, stream>>>(
          T1, L, bn2[s][bl].g, bn2[s][bl].b, bn2[s][bl].m, bn2[s][bl].v,
          WP + (size_t)(6 + s * 2 + bl) * WBLOB, T2);
      atten_kernel<<<BSZ, 256, ATT_SMEM, stream>>>(T2, L, X, inc);
      inc += 22;
    }
    bnconv1x1_kernel<<<BSZ * (L / 64), 256, 0, stream>>>(
        X, 88, L, bnt[s].g, bnt[s].b, bnt[s].m, bnt[s].v,
        WP + (size_t)(12 + s) * WBLOB, 44, T1, 0);
    pool2_kernel<<<BSZ, 256, 0, stream>>>(T1, L / 2, X);
    L /= 2;
  }
  head_kernel<<<BSZ, 64, 0, stream>>>(X, bnf.g, bnf.b, bnf.m, bnf.v, fcw, fcb,
                                      (float*)d_out);
}
